// Attention_16750372455262
// MI455X (gfx1250) — matrix operand order repaired and checked by reading, not compiled, not run on hardware
//
#include <hip/hip_runtime.h>
#include <hip/hip_bf16.h>

// ---------------------------------------------------------------------------
// Problem constants (from reference)
// ---------------------------------------------------------------------------
#define B_SZ    2
#define N_TOK   2048
#define M_TOK   64
#define L_TOK   (N_TOK + M_TOK)   // 2112
#define DIM     1024
#define HEADS   16
#define DH      64
#define INNER   (HEADS * DH)      // 1024
#define QKV_W   (3 * INNER)       // 3072
#define CLAMP_V 50.0f
#define NEG_BIG (-3.4028235e38f)

typedef __bf16 bf16;
typedef __attribute__((ext_vector_type(16))) __bf16 v16bf;
typedef __attribute__((ext_vector_type(4)))  __bf16 v4bf;
typedef __attribute__((ext_vector_type(8)))  float  v8f;
typedef __attribute__((ext_vector_type(4)))  float  v4f;
typedef __attribute__((ext_vector_type(4)))  unsigned int u32x4;
typedef __attribute__((ext_vector_type(8)))  int  i32x8;
typedef __attribute__((ext_vector_type(4)))  int  i32x4;

#if defined(__gfx1250__) && __has_builtin(__builtin_amdgcn_tensor_load_to_lds)
#define USE_TDM 1
#else
#define USE_TDM 0
#endif

#if USE_TDM
// ---------------------------------------------------------------------------
// Tensor Data Mover: 2D tile load Global -> LDS (ISA ch.8 D# descriptor).
// tile_d0 = contiguous dim (elements), tile_d1 = rows, row_stride in elements.
// Element size fixed to 2 bytes (bf16).
// ---------------------------------------------------------------------------
__device__ __forceinline__ void tdm_load_2d(unsigned lds_addr, const void* gptr,
                                            unsigned tile_d0, unsigned tile_d1,
                                            unsigned long long row_stride) {
  unsigned long long ga = (unsigned long long)(uintptr_t)gptr;
  const unsigned TD0 = 0x40000000u, TD1 = 0x40000000u;   // huge: no OOB clip
  u32x4 g0;
  g0[0] = 1u;                                        // count=1, user mode
  g0[1] = lds_addr;                                  // LDS byte address
  g0[2] = (unsigned)(ga & 0xFFFFFFFFu);              // global_addr[31:0]
  g0[3] = (unsigned)((ga >> 32) & 0x1FFFFFFu) | (2u << 30);  // [56:32] | type=2
  i32x8 g1;
  g1[0] = (int)(1u << 16);                           // mask=0, data_size=1 (2B)
  g1[1] = (int)((TD0 & 0xFFFFu) << 16);              // tensor_dim0[15:0]
  g1[2] = (int)(((TD0 >> 16) & 0xFFFFu) | ((TD1 & 0xFFFFu) << 16));
  g1[3] = (int)(((TD1 >> 16) & 0xFFFFu) | (tile_d0 << 16));   // tile_dim0
  g1[4] = (int)(tile_d1 & 0xFFFFu);                  // tile_dim1, tile_dim2=0
  g1[5] = (int)(row_stride & 0xFFFFFFFFull);         // tensor_dim0_stride[31:0]
  g1[6] = (int)((row_stride >> 32) & 0xFFFFull);     // stride[47:32], d1str=0
  g1[7] = 0;
  i32x4 z4 = {};
#if __clang_major__ >= 23
  i32x8 z8 = {};
  __builtin_amdgcn_tensor_load_to_lds(g0, g1, z4, z4, z8, 0);
#else
  __builtin_amdgcn_tensor_load_to_lds(g0, g1, z4, z4, 0);
#endif
}
#endif

__device__ __forceinline__ v8f wmma_bf16(v16bf a, v16bf b, v8f c) {
  // (neg_a, A, neg_b, B, c_mod, C, reuse_a, reuse_b)
  return __builtin_amdgcn_wmma_f32_16x16x32_bf16(false, a, false, b, (short)0, c, false, false);
}

// ---------------------------------------------------------------------------
// Fragment loaders (wave32), contiguous k order: element s of lane half h holds k = 16h + s
// ---------------------------------------------------------------------------
__device__ __forceinline__ v16bf load_a_frag(const bf16* A, int lda, int lane) {
  int m  = lane & 15;
  int kb = (lane >> 4) * 16;
  v16bf f;
#pragma unroll
  for (int v = 0; v < 8; ++v) {
    int k = kb + 2 * v;
    f[2 * v]     = A[m * lda + k];
    f[2 * v + 1] = A[m * lda + k + 1];
  }
  return f;
}

// B stored transposed in LDS as Bt[n][k] (k contiguous).
__device__ __forceinline__ v16bf load_b_frag_nk(const bf16* Bt, int ldb, int lane) {
  int n  = lane & 15;
  int kb = (lane >> 4) * 16;
  v16bf f;
#pragma unroll
  for (int v = 0; v < 8; ++v) {
    int k = kb + 2 * v;
    f[2 * v]     = Bt[n * ldb + k];
    f[2 * v + 1] = Bt[n * ldb + k + 1];
  }
  return f;
}

// B stored natural in LDS as Bm[k][n] (n contiguous).
__device__ __forceinline__ v16bf load_b_frag_kn(const bf16* Bm, int ldb, int lane) {
  int n  = lane & 15;
  int kb = (lane >> 4) * 16;
  v16bf f;
#pragma unroll
  for (int v = 0; v < 8; ++v) {
    int k = kb + 2 * v;
    f[2 * v]     = Bm[k * ldb + n];
    f[2 * v + 1] = Bm[(k + 1) * ldb + n];
  }
  return f;
}

// ---------------------------------------------------------------------------
// Kernel 1: RMSNorm (f32 in) -> bf16 combined activations Xn[B, L, DIM]
// ---------------------------------------------------------------------------
__global__ __launch_bounds__(256) void rmsnorm_kernel(
    const float* __restrict__ ms, const float* __restrict__ ac,
    const float* __restrict__ w,  const float* __restrict__ wa,
    bf16* __restrict__ Xn) {
  int rowg = blockIdx.x;                 // 0 .. B*L-1
  int b = rowg / L_TOK;
  int l = rowg - b * L_TOK;
  const float* x; const float* ww;
  if (l < N_TOK) { x = ms + ((size_t)b * N_TOK + l) * DIM;           ww = w;  }
  else           { x = ac + ((size_t)b * M_TOK + (l - N_TOK)) * DIM; ww = wa; }

  int i4 = threadIdx.x * 4;
  v4f xv = *(const v4f*)(x + i4);
  float ss = xv[0] * xv[0] + xv[1] * xv[1] + xv[2] * xv[2] + xv[3] * xv[3];

  __shared__ float red[256];
  red[threadIdx.x] = ss;
  __syncthreads();
#pragma unroll
  for (int s = 128; s > 0; s >>= 1) {
    if (threadIdx.x < s) red[threadIdx.x] += red[threadIdx.x + s];
    __syncthreads();
  }
  float inv = rsqrtf(red[0] * (1.0f / DIM) + 1.1920929e-07f);

  v4f wv = *(const v4f*)(ww + i4);
  v4bf o;
#pragma unroll
  for (int e = 0; e < 4; ++e) o[e] = (bf16)(xv[e] * inv * wv[e]);
  *(v4bf*)(Xn + (size_t)rowg * DIM + i4) = o;
}

// ---------------------------------------------------------------------------
// Kernel 2: f32 -> bf16 elementwise (weights), 4-wide
// ---------------------------------------------------------------------------
__global__ void f2bf_kernel(const float* __restrict__ src, bf16* __restrict__ dst, int n4) {
  int i = blockIdx.x * blockDim.x + threadIdx.x;
  if (i < n4) {
    v4f s = *(const v4f*)(src + (size_t)i * 4);
    v4bf o;
#pragma unroll
    for (int e = 0; e < 4; ++e) o[e] = (bf16)s[e];
    *(v4bf*)(dst + (size_t)i * 4) = o;
  }
}

// ---------------------------------------------------------------------------
// Kernel 3: QKV GEMM.  Block tile 64x128, 8 waves as 2x4, wave tile 32x32.
// A tile staged via TDM (wave 0), B tile transposed by all threads.
// ---------------------------------------------------------------------------
#define BK  32
#define LDB (BK + 4)
__global__ __launch_bounds__(256) void qkv_gemm_kernel(
    const bf16* __restrict__ Xn, const bf16* __restrict__ Wq,
    const bf16* __restrict__ Wa,
    bf16* __restrict__ Qd, bf16* __restrict__ Kd, bf16* __restrict__ Vd) {
  __shared__ bf16 As[64][BK];     // dense: TDM destination
  __shared__ bf16 Bs[128][LDB];   // transposed: Bs[n][k]
  int b    = blockIdx.z;
  int row0 = blockIdx.y * 64;       // tiles never straddle the N boundary
  int col0 = blockIdx.x * 128;      // tiles never straddle a part boundary
  const bf16* W    = (row0 < N_TOK) ? Wq : Wa;
  const bf16* Arow = Xn + ((size_t)b * L_TOK + row0) * DIM;

  int tid = threadIdx.x, wave = tid >> 5, lane = tid & 31;
  int wm = (wave & 1) * 32;
  int wn = (wave >> 1) * 32;
#if USE_TDM
  unsigned as_lds = (unsigned)(uintptr_t)&As[0][0];
#endif

  v8f acc[2][2] = {};
  for (int k0 = 0; k0 < DIM; k0 += BK) {
#if USE_TDM
    if (wave == 0) {
      tdm_load_2d(as_lds, Arow + k0, BK, 64, (unsigned long long)DIM);
      __builtin_amdgcn_s_wait_tensorcnt(0);
    }
#else
#pragma unroll
    for (int t = 0; t < 2; ++t) {
      int v = tid + t * 256;
      int r = v >> 3, c = (v & 7) * 4;
      *(v4bf*)(&As[r][c]) = *(const v4bf*)(Arow + (size_t)r * DIM + k0 + c);
    }
#endif
    // B: 32k x 128n bf16 = 1024 x v4bf along n, 4 per thread, transpose into LDS.
#pragma unroll
    for (int t = 0; t < 4; ++t) {
      int v = tid + t * 256;
      int kk = v >> 5, nn = (v & 31) * 4;
      v4bf bv = *(const v4bf*)(W + (size_t)(k0 + kk) * QKV_W + col0 + nn);
#pragma unroll
      for (int e = 0; e < 4; ++e) Bs[nn + e][kk] = bv[e];
    }
    // Prefetch next k-tile (global_prefetch_b8).
    if (k0 + BK < DIM) {
      __builtin_prefetch(W + (size_t)(k0 + BK + (tid >> 5)) * QKV_W + col0, 0, 1);
    }
    __syncthreads();
    v16bf a0 = load_a_frag(&As[wm][0],       BK, lane);
    v16bf a1 = load_a_frag(&As[wm + 16][0],  BK, lane);
    v16bf b0 = load_b_frag_nk(&Bs[wn][0],      LDB, lane);
    v16bf b1 = load_b_frag_nk(&Bs[wn + 16][0], LDB, lane);
    acc[0][0] = wmma_bf16(a0, b0, acc[0][0]);
    acc[0][1] = wmma_bf16(a0, b1, acc[0][1]);
    acc[1][0] = wmma_bf16(a1, b0, acc[1][0]);
    acc[1][1] = wmma_bf16(a1, b1, acc[1][1]);
    __syncthreads();
  }

  int lane_n = lane & 15, half = lane >> 4;
  int part = col0 >> 10;            // block-uniform: 0=Q 1=K 2=V
  size_t bH = (size_t)b * HEADS;
  if (part == 0) {
#pragma unroll
    for (int mi = 0; mi < 2; ++mi)
#pragma unroll
      for (int ni = 0; ni < 2; ++ni)
#pragma unroll
        for (int r = 0; r < 8; ++r) {
          int m  = row0 + wm + mi * 16 + r + 8 * half;
          int cc = (col0 & (INNER - 1)) + wn + ni * 16 + lane_n;
          int h = cc >> 6, d = cc & 63;
          Qd[((bH + h) * L_TOK + m) * DH + d] = (bf16)(acc[mi][ni][r] * 0.125f);
        }
  } else if (part == 1) {
#pragma unroll
    for (int mi = 0; mi < 2; ++mi)
#pragma unroll
      for (int ni = 0; ni < 2; ++ni)
#pragma unroll
        for (int r = 0; r < 8; ++r) {
          int m  = row0 + wm + mi * 16 + r + 8 * half;
          int cc = (col0 & (INNER - 1)) + wn + ni * 16 + lane_n;
          int h = cc >> 6, d = cc & 63;
          Kd[((bH + h) * DH + d) * L_TOK + m] = (bf16)acc[mi][ni][r];
        }
  } else {
#pragma unroll
    for (int mi = 0; mi < 2; ++mi)
#pragma unroll
      for (int ni = 0; ni < 2; ++ni)
#pragma unroll
        for (int r = 0; r < 8; ++r) {
          int m  = row0 + wm + mi * 16 + r + 8 * half;
          int cc = (col0 & (INNER - 1)) + wn + ni * 16 + lane_n;
          int h = cc >> 6, d = cc & 63;
          Vd[((bH + h) * L_TOK + m) * DH + d] = (bf16)acc[mi][ni][r];
        }
  }
}

// ---------------------------------------------------------------------------
// Kernel 4: flash attention with tanh soft-clamp and prefix mask.
// One wave per (b, h, 16-query block).  32-key tiles via TDM, online softmax.
// ---------------------------------------------------------------------------
__global__ __launch_bounds__(32) void attn_kernel(
    const bf16* __restrict__ Qd, const bf16* __restrict__ Kd,
    const bf16* __restrict__ Vd, bf16* __restrict__ Att) {
  __shared__ bf16 Kt[64][32];   // [d][j]  dense: TDM destination
  __shared__ bf16 Vs[32][64];   // [j][d]  dense: TDM destination
  __shared__ bf16 Ps[16][36];   // [m][j]  P re-layout staging

  int b = blockIdx.z, h = blockIdx.y;
  int q0 = blockIdx.x * 16;
  int lane = threadIdx.x;
  size_t bh = (size_t)(b * HEADS + h);
  const bf16* Qp = Qd + (bh * L_TOK + q0) * DH;
  const bf16* Kp = Kd + bh * (size_t)DH * L_TOK;   // [d][l]
  const bf16* Vp = Vd + bh * (size_t)L_TOK * DH;   // [l][d]
#if USE_TDM
  unsigned kt_lds = (unsigned)(uintptr_t)&Kt[0][0];
  unsigned vs_lds = (unsigned)(uintptr_t)&Vs[0][0];
#endif

  // Q A-fragments (K-dim 64 -> two frags), loaded once, already scaled.
  v16bf qa0, qa1;
  {
    int m = lane & 15, kb = (lane >> 4) * 16;
#pragma unroll
    for (int v = 0; v < 8; ++v) {
      int k = kb + 2 * v;
      qa0[2 * v]     = Qp[m * DH + k];
      qa0[2 * v + 1] = Qp[m * DH + k + 1];
      qa1[2 * v]     = Qp[m * DH + 32 + k];
      qa1[2 * v + 1] = Qp[m * DH + 32 + k + 1];
    }
  }

  v8f o0 = {}, o1 = {}, o2 = {}, o3 = {};
  float mrow[8], lrow[8];
#pragma unroll
  for (int r = 0; r < 8; ++r) { mrow[r] = NEG_BIG; lrow[r] = 0.0f; }

  int half = lane >> 4, lane_n = lane & 15;

  for (int j0 = 0; j0 < L_TOK; j0 += 32) {
    // Causal skip: tile fully masked if all j > all i and entire tile < N.
    if ((j0 > q0 + 15) && (j0 + 31 < N_TOK)) continue;

#if USE_TDM
    // Async tensor DMA: K tile [64 x 32] (row stride L), V tile [32 x 64].
    tdm_load_2d(kt_lds, Kp + j0, 32, 64, (unsigned long long)L_TOK);
    tdm_load_2d(vs_lds, Vp + (size_t)j0 * DH, 64, 32, (unsigned long long)DH);
    __builtin_amdgcn_s_wait_tensorcnt(0);
#else
#pragma unroll
    for (int t = 0; t < 16; ++t) {           // 64*32/4 = 512 v4bf / 32 lanes
      int v = lane + t * 32;
      int d = v >> 3, jj = (v & 7) * 4;
      *(v4bf*)(&Kt[d][jj]) = *(const v4bf*)(Kp + (size_t)d * L_TOK + j0 + jj);
      int jj2 = v >> 4, dd = (v & 15) * 4;
      *(v4bf*)(&Vs[jj2][dd]) = *(const v4bf*)(Vp + (size_t)(j0 + jj2) * DH + dd);
    }
#endif
    // Speculative prefetch of the next key tile.
    __builtin_prefetch(Kp + (size_t)(lane << 1) * L_TOK + j0 + 32, 0, 1);
    __builtin_prefetch(Vp + (size_t)(j0 + 32 + lane) * DH, 0, 1);
    __syncthreads();

    // S = Q * K^T  (16 x 32 as two 16x16 accumulators, 2 K-steps each)
    v8f s0 = {}, s1 = {};
    s0 = wmma_bf16(qa0, load_b_frag_kn(&Kt[0][0],   32, lane), s0);
    s0 = wmma_bf16(qa1, load_b_frag_kn(&Kt[32][0],  32, lane), s0);
    s1 = wmma_bf16(qa0, load_b_frag_kn(&Kt[0][16],  32, lane), s1);
    s1 = wmma_bf16(qa1, load_b_frag_kn(&Kt[32][16], 32, lane), s1);

    // Soft-clamp, mask, online softmax update (C layout: row = r + 8*half).
#pragma unroll
    for (int r = 0; r < 8; ++r) {
      int i_abs = q0 + r + 8 * half;
      int jlo = j0 + lane_n, jhi = jlo + 16;
      float v0 = CLAMP_V * tanhf(s0[r] * (1.0f / CLAMP_V));
      float v1 = CLAMP_V * tanhf(s1[r] * (1.0f / CLAMP_V));
      if ((jlo > i_abs) && (jlo < N_TOK)) v0 = NEG_BIG;
      if ((jhi > i_abs) && (jhi < N_TOK)) v1 = NEG_BIG;

      float rmax = fmaxf(v0, v1);
#pragma unroll
      for (int off = 8; off; off >>= 1) rmax = fmaxf(rmax, __shfl_xor(rmax, off, 32));
      float m_new = fmaxf(mrow[r], rmax);
      float alpha = __expf(mrow[r] - m_new);
      float p0 = (v0 > -1.0e37f) ? __expf(v0 - m_new) : 0.0f;
      float p1 = (v1 > -1.0e37f) ? __expf(v1 - m_new) : 0.0f;
      float psum = p0 + p1;
#pragma unroll
      for (int off = 8; off; off >>= 1) psum += __shfl_xor(psum, off, 32);
      lrow[r] = lrow[r] * alpha + psum;
      mrow[r] = m_new;
      o0[r] *= alpha; o1[r] *= alpha; o2[r] *= alpha; o3[r] *= alpha;

      Ps[r + 8 * half][lane_n]      = (bf16)p0;
      Ps[r + 8 * half][lane_n + 16] = (bf16)p1;
    }
    __syncthreads();

    // O += P * V  (A = P 16x32 via LDS re-layout; B = V tile, 4 column groups)
    v16bf pa = load_a_frag(&Ps[0][0], 36, lane);
    o0 = wmma_bf16(pa, load_b_frag_kn(&Vs[0][0],  64, lane), o0);
    o1 = wmma_bf16(pa, load_b_frag_kn(&Vs[0][16], 64, lane), o1);
    o2 = wmma_bf16(pa, load_b_frag_kn(&Vs[0][32], 64, lane), o2);
    o3 = wmma_bf16(pa, load_b_frag_kn(&Vs[0][48], 64, lane), o3);
    __syncthreads();
  }

  // Normalize and store to Att[b, l, h*DH + d] (bf16).
#pragma unroll
  for (int r = 0; r < 8; ++r) {
    float inv = 1.0f / lrow[r];
    size_t base = ((size_t)b * L_TOK + q0 + r + 8 * half) * INNER + h * DH;
    Att[base + lane_n]      = (bf16)(o0[r] * inv);
    Att[base + 16 + lane_n] = (bf16)(o1[r] * inv);
    Att[base + 32 + lane_n] = (bf16)(o2[r] * inv);
    Att[base + 48 + lane_n] = (bf16)(o3[r] * inv);
  }
}

// ---------------------------------------------------------------------------
// Kernel 5: output projection.  Block tile 64x128, wave tile 32x32.
// ---------------------------------------------------------------------------
__global__ __launch_bounds__(256) void outproj_kernel(
    const bf16* __restrict__ Att, const bf16* __restrict__ Wo,
    const bf16* __restrict__ Wao, float* __restrict__ out1,
    float* __restrict__ out2) {
  __shared__ bf16 As[64][BK];     // dense: TDM destination
  __shared__ bf16 Bs[128][LDB];
  int b    = blockIdx.z;
  int row0 = blockIdx.y * 64;
  int col0 = blockIdx.x * 128;
  const bf16* W    = (row0 < N_TOK) ? Wo : Wao;
  const bf16* Arow = Att + ((size_t)b * L_TOK + row0) * INNER;

  int tid = threadIdx.x, wave = tid >> 5, lane = tid & 31;
  int wm = (wave & 1) * 32;
  int wn = (wave >> 1) * 32;
#if USE_TDM
  unsigned as_lds = (unsigned)(uintptr_t)&As[0][0];
#endif

  v8f acc[2][2] = {};
  for (int k0 = 0; k0 < INNER; k0 += BK) {
#if USE_TDM
    if (wave == 0) {
      tdm_load_2d(as_lds, Arow + k0, BK, 64, (unsigned long long)INNER);
      __builtin_amdgcn_s_wait_tensorcnt(0);
    }
#else
#pragma unroll
    for (int t = 0; t < 2; ++t) {
      int v = tid + t * 256;
      int r = v >> 3, c = (v & 7) * 4;
      *(v4bf*)(&As[r][c]) = *(const v4bf*)(Arow + (size_t)r * INNER + k0 + c);
    }
#endif
#pragma unroll
    for (int t = 0; t < 4; ++t) {
      int v = tid + t * 256;
      int kk = v >> 5, nn = (v & 31) * 4;
      v4bf bv = *(const v4bf*)(W + (size_t)(k0 + kk) * DIM + col0 + nn);
#pragma unroll
      for (int e = 0; e < 4; ++e) Bs[nn + e][kk] = bv[e];
    }
    if (k0 + BK < INNER) {
      __builtin_prefetch(W + (size_t)(k0 + BK + (tid >> 5)) * DIM + col0, 0, 1);
    }
    __syncthreads();
    v16bf a0 = load_a_frag(&As[wm][0],      BK, lane);
    v16bf a1 = load_a_frag(&As[wm + 16][0], BK, lane);
    v16bf b0 = load_b_frag_nk(&Bs[wn][0],      LDB, lane);
    v16bf b1 = load_b_frag_nk(&Bs[wn + 16][0], LDB, lane);
    acc[0][0] = wmma_bf16(a0, b0, acc[0][0]);
    acc[0][1] = wmma_bf16(a0, b1, acc[0][1]);
    acc[1][0] = wmma_bf16(a1, b0, acc[1][0]);
    acc[1][1] = wmma_bf16(a1, b1, acc[1][1]);
    __syncthreads();
  }

  int lane_n = lane & 15, half = lane >> 4;
  if (row0 < N_TOK) {
#pragma unroll
    for (int mi = 0; mi < 2; ++mi)
#pragma unroll
      for (int ni = 0; ni < 2; ++ni)
#pragma unroll
        for (int r = 0; r < 8; ++r) {
          int m = row0 + wm + mi * 16 + r + 8 * half;
          int c = col0 + wn + ni * 16 + lane_n;
          out1[((size_t)b * N_TOK + m) * DIM + c] = acc[mi][ni][r];
        }
  } else {
#pragma unroll
    for (int mi = 0; mi < 2; ++mi)
#pragma unroll
      for (int ni = 0; ni < 2; ++ni)
#pragma unroll
        for (int r = 0; r < 8; ++r) {
          int m = row0 + wm + mi * 16 + r + 8 * half - N_TOK;
          int c = col0 + wn + ni * 16 + lane_n;
          out2[((size_t)b * M_TOK + m) * DIM + c] = acc[mi][ni][r];
        }
  }
}

// ---------------------------------------------------------------------------
// Host launcher
// ---------------------------------------------------------------------------
extern "C" void kernel_launch(void* const* d_in, const int* in_sizes, int n_in,
                              void* d_out, int out_size, void* d_ws, size_t ws_size,
                              hipStream_t stream) {
  const float* ms    = (const float*)d_in[0];
  const float* ac    = (const float*)d_in[1];
  const float* rw    = (const float*)d_in[2];
  const float* rwa   = (const float*)d_in[3];
  const float* wqkv  = (const float*)d_in[4];
  const float* wout  = (const float*)d_in[5];
  const float* waqkv = (const float*)d_in[6];
  const float* waout = (const float*)d_in[7];
  float* out = (float*)d_out;

  // Workspace carve-up (~60 MB total, 256B aligned).
  char*  ws  = (char*)d_ws;
  size_t off = 0;
  auto alloc = [&](size_t bytes) -> void* {
    void* p = ws + off;
    off += (bytes + 255) & ~(size_t)255;
    return p;
  };
  const size_t actE = (size_t)B_SZ * L_TOK * DIM;          // 4,325,376
  const size_t hdE  = (size_t)B_SZ * HEADS * L_TOK * DH;   // 4,325,376
  bf16* xn      = (bf16*)alloc(actE * 2);
  bf16* wqkv_b  = (bf16*)alloc((size_t)DIM * QKV_W * 2);
  bf16* waqkv_b = (bf16*)alloc((size_t)DIM * QKV_W * 2);
  bf16* wout_b  = (bf16*)alloc((size_t)INNER * DIM * 2);
  bf16* waout_b = (bf16*)alloc((size_t)INNER * DIM * 2);
  bf16* q       = (bf16*)alloc(hdE * 2);
  bf16* k       = (bf16*)alloc(hdE * 2);
  bf16* v       = (bf16*)alloc(hdE * 2);
  bf16* att     = (bf16*)alloc(actE * 2);

  // 1) RMSNorm -> bf16 activations
  rmsnorm_kernel<<<B_SZ * L_TOK, 256, 0, stream>>>(ms, ac, rw, rwa, xn);

  // 2) Weight conversions (4-wide)
  int nqkv4 = DIM * QKV_W / 4, nout4 = INNER * DIM / 4;
  f2bf_kernel<<<(nqkv4 + 255) / 256, 256, 0, stream>>>(wqkv,  wqkv_b,  nqkv4);
  f2bf_kernel<<<(nqkv4 + 255) / 256, 256, 0, stream>>>(waqkv, waqkv_b, nqkv4);
  f2bf_kernel<<<(nout4 + 255) / 256, 256, 0, stream>>>(wout,  wout_b,  nout4);
  f2bf_kernel<<<(nout4 + 255) / 256, 256, 0, stream>>>(waout, waout_b, nout4);

  // 3) QKV projection
  qkv_gemm_kernel<<<dim3(QKV_W / 128, L_TOK / 64, B_SZ), 256, 0, stream>>>(
      xn, wqkv_b, waqkv_b, q, k, v);

  // 4) Attention
  attn_kernel<<<dim3(L_TOK / 16, HEADS, B_SZ), 32, 0, stream>>>(q, k, v, att);

  // 5) Output projections -> f32 outputs (concatenated)
  float* out2 = out + (size_t)B_SZ * N_TOK * DIM;
  outproj_kernel<<<dim3(DIM / 128, L_TOK / 64, B_SZ), 256, 0, stream>>>(
      att, wout_b, waout_b, out, out2);
}
